// RGCN_16904991277355
// MI455X (gfx1250) — matrix operand order repaired and checked by reading, not compiled, not run on hardware
//
#include <hip/hip_runtime.h>
#include <hip/hip_bf16.h>

// ---------------------------------------------------------------------------
// RGCN (2-layer) for MI455X / gfx1250, wave32 + WMMA.
// Strategy: per-relation mean is linear -> aggregate-then-transform.
//   agg[i,r,:] = sum_{(j->i) in rel r} x[j,:]     (f32 scatter-add atomics)
//   out[i,:]   = [x[i,:] | agg[i,0,:]/c .. agg[i,7,:]/c] @ [root; W0..W7] + b
// GEMM: M-tile = 16 nodes per block, K = 9*128 = 1152 staged in LDS as f16,
// one wave per 16-col slab, v_wmma_f32_16x16x32_f16 with f32 accumulation.
// ---------------------------------------------------------------------------

#define FI     128              // feature width of both GEMM inputs
#define RNUM   8                // relations
#define KTOT   ((RNUM + 1) * FI)   // 1152
#define KSTEPS (KTOT / 32)         // 36
#define APAD   8                   // f16 pad -> row stride 580 dwords (no bank conflicts)
#define AROW   (KTOT + APAD)

typedef __attribute__((ext_vector_type(16))) _Float16 v16h;
typedef __attribute__((ext_vector_type(8)))  _Float16 v8h;
typedef __attribute__((ext_vector_type(8)))  float    v8f;

// -------------------------------- utility ----------------------------------

__global__ void zero_f32(float* __restrict__ p, size_t n) {
  size_t i = (size_t)blockIdx.x * blockDim.x + threadIdx.x;
  size_t stride = (size_t)gridDim.x * blockDim.x;
  for (; i < n; i += stride) p[i] = 0.0f;
}

// per-(dst, relation) in-degree, float (graph-only; computed once)
__global__ void count_edges(const int* __restrict__ dst,
                            const int* __restrict__ et,
                            float* __restrict__ cnt, int E) {
  int e = blockIdx.x * blockDim.x + threadIdx.x;
  if (e < E) unsafeAtomicAdd(&cnt[dst[e] * RNUM + et[e]], 1.0f);
}

// one wave per edge: 32 lanes x float4 = 128 features, coalesced gather,
// f32 scatter-add (global_atomic_add_f32) into agg[dst, r, :]
__global__ __launch_bounds__(256)
void aggregate_edges(const float* __restrict__ feat,
                     const int* __restrict__ src,
                     const int* __restrict__ dst,
                     const int* __restrict__ et,
                     float* __restrict__ agg, int E) {
  int gtid = blockIdx.x * blockDim.x + threadIdx.x;
  int e = gtid >> 5;
  int lane = gtid & 31;
  if (e >= E) return;
  int s = src[e], d = dst[e], r = et[e];
  const float4 v = ((const float4*)(feat + (size_t)s * FI))[lane];
  float* ob = agg + ((size_t)d * RNUM + r) * FI + lane * 4;
  unsafeAtomicAdd(ob + 0, v.x);
  unsafeAtomicAdd(ob + 1, v.y);
  unsafeAtomicAdd(ob + 2, v.z);
  unsafeAtomicAdd(ob + 3, v.w);
}

// Pack [root; W0..W7] (f32, row-major [KTOT, Fo]) into the WMMA B-fragment
// per-lane order: bpack[(((slab*KSTEPS + ks)*32 + lane)*16 + e]
// B 32x16 order used here (contiguous k, same as the A reads): N = slab*16 + lane%16, K = ks*32 + 16*(lane/16) + e
__global__ void pack_weights(const float* __restrict__ root,   // [FI,  Fo]
                             const float* __restrict__ W,      // [R*FI, Fo]
                             _Float16* __restrict__ bpack, int Fo) {
  int o = blockIdx.x * blockDim.x + threadIdx.x;
  if (o >= KTOT * Fo) return;
  int e     = o & 15;
  int lane  = (o >> 4) & 31;
  int rest  = o >> 9;
  int ks    = rest % KSTEPS;
  int slab  = rest / KSTEPS;
  int k     = ks * 32 + ((lane >> 4) << 4) + e;
  int col   = slab * 16 + (lane & 15);
  float v   = (k < FI) ? root[k * Fo + col] : W[(size_t)(k - FI) * Fo + col];
  bpack[o]  = (_Float16)v;
}

// ------------------------------- WMMA GEMM ---------------------------------
// block = (Fo/16) waves; grid.x = N/16 (N = 50000 = 3125*16, no tail).
__global__ __launch_bounds__(256)
void rgcn_gemm(const float* __restrict__ feat,     // [N, FI]
               const float* __restrict__ agg,      // [N, RNUM, FI]
               const float* __restrict__ cnt,      // [N, RNUM]
               const _Float16* __restrict__ bpack, // packed B fragments
               const float* __restrict__ bias,     // [Fo]
               float* __restrict__ out,            // [N, Fo]
               int Fo, int do_relu) {
  __shared__ __align__(16) _Float16 Alds[16][AROW];
  __shared__ float sscale[16][RNUM];

  const int tid   = threadIdx.x;
  const int node0 = blockIdx.x * 16;

  if (tid < 16 * RNUM) {
    int m = tid / RNUM, r = tid % RNUM;
    sscale[m][r] = 1.0f / fmaxf(cnt[(size_t)(node0 + m) * RNUM + r], 1.0f);
  }
  __syncthreads();

  // stage A-slab: [x | agg/c] for 16 nodes, f32 -> f16, row-major in LDS
  for (int idx = tid; idx < 16 * KTOT; idx += blockDim.x) {
    int m = idx / KTOT;
    int k = idx - m * KTOT;
    int node = node0 + m;
    float v;
    if (k < FI) {
      v = feat[(size_t)node * FI + k];
    } else {
      int r  = (k - FI) >> 7;     // / 128
      int kl = k & (FI - 1);
      v = agg[((size_t)node * RNUM + r) * FI + kl] * sscale[m][r];
    }
    Alds[m][k] = (_Float16)v;
  }
  __syncthreads();

  const int lane = tid & 31;
  const int wave = tid >> 5;      // output column slab
  const int row  = lane & 15;     // A matrix row (M)
  const int hi   = lane >> 4;     // lane half selects K sub-range

  v8f acc = {};
  const _Float16* bp = bpack + ((size_t)wave * KSTEPS * 32 + lane) * 16;

  for (int ks = 0; ks < KSTEPS; ++ks) {
    // A 16x32 layout (matches B pack): elems 0..15 = K koff..koff+15
    const int koff = ks * 32 + hi * 16;
    const _Float16* ap = &Alds[row][koff];
    v8h alo = *(const v8h*)(ap);
    v8h ahi = *(const v8h*)(ap + 8);
    v16h a = __builtin_shufflevector(alo, ahi, 0, 1, 2, 3, 4, 5, 6, 7,
                                               8, 9, 10, 11, 12, 13, 14, 15);
    v16h b = *(const v16h*)(bp + (size_t)ks * 512);   // 32B -> 2x global_load_b128
    acc = __builtin_amdgcn_wmma_f32_16x16x32_f16(
        /*neg_a=*/false, a, /*neg_b=*/false, b,
        /*c_mod=*/(short)0, acc, /*reuse_a=*/false, /*reuse_b=*/false);
  }

  // C/D layout: VGPR v -> M = v + 8*(lane/16); N = lane%16
  const int col = wave * 16 + (lane & 15);
  const float bv = bias[col];
#pragma unroll
  for (int v = 0; v < 8; ++v) {
    int m = v + hi * 8;
    float o = acc[v] + bv;
    if (do_relu) o = fmaxf(o, 0.0f);
    out[(size_t)(node0 + m) * Fo + col] = o;
  }
}

// ------------------------------- launcher ----------------------------------

extern "C" void kernel_launch(void* const* d_in, const int* in_sizes, int n_in,
                              void* d_out, int out_size, void* d_ws, size_t ws_size,
                              hipStream_t stream) {
  const float* x     = (const float*)d_in[0];
  const int*   eidx  = (const int*)d_in[1];
  const int*   etype = (const int*)d_in[2];
  const float* W1    = (const float*)d_in[3];
  const float* root1 = (const float*)d_in[4];
  const float* b1    = (const float*)d_in[5];
  const float* W2    = (const float*)d_in[6];
  const float* root2 = (const float*)d_in[7];
  const float* b2    = (const float*)d_in[8];

  const int N = in_sizes[0] / FI;   // 50000 (divisible by 16)
  const int E = in_sizes[2];        // 800000
  const int* src = eidx;
  const int* dst = eidx + E;
  const int FO1 = 128, FO2 = 64;

  // workspace carve (256B aligned): agg 204.8MB, cnt 1.6MB, h 25.6MB, packs
  uint8_t* ws = (uint8_t*)d_ws;
  auto carve = [&](size_t bytes) -> void* {
    void* p = (void*)ws;
    ws += (bytes + 255) & ~(size_t)255;
    return p;
  };
  float*    agg = (float*)carve((size_t)N * RNUM * FI * sizeof(float));
  float*    cnt = (float*)carve((size_t)N * RNUM * sizeof(float));
  float*    h   = (float*)carve((size_t)N * FI * sizeof(float));
  _Float16* bp1 = (_Float16*)carve((size_t)KTOT * FO1 * sizeof(_Float16));
  _Float16* bp2 = (_Float16*)carve((size_t)KTOT * FO2 * sizeof(_Float16));

  const size_t nAgg = (size_t)N * RNUM * FI;
  const size_t nCnt = (size_t)N * RNUM;

  // graph-only degree counts (shared by both layers)
  zero_f32<<<2048, 256, 0, stream>>>(cnt, nCnt);
  count_edges<<<(E + 255) / 256, 256, 0, stream>>>(dst, etype, cnt, E);

  // pre-pack weights into WMMA B-fragment order
  pack_weights<<<(KTOT * FO1 + 255) / 256, 256, 0, stream>>>(root1, W1, bp1, FO1);
  pack_weights<<<(KTOT * FO2 + 255) / 256, 256, 0, stream>>>(root2, W2, bp2, FO2);

  // ---- layer 1: x -> h (ReLU) ----
  zero_f32<<<4096, 256, 0, stream>>>(agg, nAgg);
  aggregate_edges<<<(E * 32 + 255) / 256, 256, 0, stream>>>(x, src, dst, etype, agg, E);
  rgcn_gemm<<<N / 16, (FO1 / 16) * 32, 0, stream>>>(x, agg, cnt, bp1, b1, h, FO1, 1);

  // ---- layer 2: h -> out ----
  zero_f32<<<4096, 256, 0, stream>>>(agg, nAgg);
  aggregate_edges<<<(E * 32 + 255) / 256, 256, 0, stream>>>(h, src, dst, etype, agg, E);
  rgcn_gemm<<<N / 16, (FO2 / 16) * 32, 0, stream>>>(h, agg, cnt, bp2, b2,
                                                    (float*)d_out, FO2, 0);
}
